// MoELoRA_21260088115991
// MI455X (gfx1250) — compile-verified, matrix operand order checked by reading, not run on hardware
//
#include <hip/hip_runtime.h>
#include <math.h>

// ---------------- problem constants ----------------
#define D_    1024
#define E_    8
#define R_    64
#define HK    (E_ * R_)        // 512, concatenated-expert K for fc2
#define MT    64               // tokens per block
#define KC    128              // K-chunk staged in LDS
#define XS_STRIDE (KC + 8)     // 136 bf16 (pad to stagger banks, keeps 16B align)
#define HS_STRIDE (HK + 8)     // 520 bf16

typedef __attribute__((ext_vector_type(16))) __bf16 v16bf;
typedef __attribute__((ext_vector_type(8)))  float  v8f;

#define WMMA_BF16(A, B, C) \
  __builtin_amdgcn_wmma_f32_16x16x32_bf16(false, (A), false, (B), (short)0, (C), false, false)

// Load a 16x32 bf16 WMMA operand fragment (A-layout; B loaded from B^T rows).
// Per ISA: lane L holds row (L&15), K elems {k0..k0+7, k0+16..k0+23}, k0=(L>>4)*8.
// Two 16-byte loads per lane; requires 16B-aligned base and stride%8==0.
__device__ __forceinline__ v16bf load_frag(const __bf16* base, int stride) {
  const int lane = threadIdx.x & 31;
  const __bf16* p = base + (lane & 15) * stride + ((lane >> 4) << 3);
  union { v16bf v; uint4 u[2]; } t;
  t.u[0] = *(const uint4*)(p);
  t.u[1] = *(const uint4*)(p + 16);
  return t.v;
}

__device__ __forceinline__ void store4bf(__bf16* dst, float4 v) {
  __align__(8) __bf16 q[4] = { (__bf16)v.x, (__bf16)v.y, (__bf16)v.z, (__bf16)v.w };
  *(uint2*)dst = *(const uint2*)q;
}

// --------- prep: W1 -> bf16, W2 [E,D,R] -> [D, E*R] bf16, Wg padded to 16 rows ---------
__global__ void moe_lora_prep(const float* __restrict__ Wg,
                              const float* __restrict__ W1,
                              const float* __restrict__ W2,
                              __bf16* __restrict__ W1bf,
                              __bf16* __restrict__ W2r,
                              __bf16* __restrict__ Wgbf) {
  const int idx = blockIdx.x * blockDim.x + threadIdx.x;
  if (idx < E_ * R_ * D_) W1bf[idx] = (__bf16)W1[idx];
  if (idx < D_ * HK) {
    const int d  = idx >> 9;          // / 512
    const int er = idx & (HK - 1);
    const int e  = er >> 6;
    const int r  = er & (R_ - 1);
    W2r[idx] = (__bf16)W2[((size_t)e * D_ + d) * R_ + r];
  }
  if (idx < 16 * D_) {
    const int row = idx >> 10;
    const int c   = idx & (D_ - 1);
    Wgbf[idx] = (row < E_) ? (__bf16)Wg[row * D_ + c] : (__bf16)(0.0f);
  }
}

// --------- fused MoE-LoRA: router + top2-softmax + gelu(x@W1^T)@W2^T, dense-masked ---------
__global__ __launch_bounds__(256)
void moe_lora_main(const float* __restrict__ x,
                   const float* __restrict__ bg,
                   const __bf16* __restrict__ W1bf,
                   const __bf16* __restrict__ W2r,
                   const __bf16* __restrict__ Wgbf,
                   float* __restrict__ out) {
  extern __shared__ __align__(16) char smem[];
  __bf16* Xs0  = (__bf16*)smem;                 // [MT][XS_STRIDE] bf16, double-buffered
  __bf16* Xs1  = Xs0 + MT * XS_STRIDE;
  __bf16* Hs   = Xs1 + MT * XS_STRIDE;          // [MT][HS_STRIDE] bf16 (gate*gelu(h))
  float*  Llog = (float*)(Hs + MT * HS_STRIDE); // [MT][E] router logits
  float*  Gt   = Llog + MT * E_;                // [MT][E] dense gates (x SCALING)

  const int tid  = threadIdx.x;
  const int wave = tid >> 5;     // 0..7 -> owns expert `wave` in phase 1
  const int lane = tid & 31;
  const int hi   = (lane >> 4) << 3;
  const int tile = blockIdx.x * MT;

  const v8f vz = {0.f, 0.f, 0.f, 0.f, 0.f, 0.f, 0.f, 0.f};
  v8f hacc[4][4];
#pragma unroll
  for (int a = 0; a < 4; ++a)
#pragma unroll
    for (int b = 0; b < 4; ++b) hacc[a][b] = vz;
  v8f racc = vz;

  // ---- prefetch chunk 0 of x into registers, convert to bf16 in LDS ----
  float4 xr[8];
#pragma unroll
  for (int i = 0; i < 8; ++i) {
    const int f = tid + i * 256;                 // float4 id: row=f>>5, col4=f&31
    xr[i] = *(const float4*)(x + (size_t)(tile + (f >> 5)) * D_ + ((f & 31) << 2));
  }
#pragma unroll
  for (int i = 0; i < 8; ++i) {
    const int f = tid + i * 256;
    store4bf(Xs0 + (f >> 5) * XS_STRIDE + ((f & 31) << 2), xr[i]);
  }
  __syncthreads();

  const __bf16* w1e = W1bf + (size_t)wave * (R_ * D_);

  // ---- phase 1: h[e=wave] = x @ W1[e]^T  (+ router logits on waves 0-3) ----
#pragma unroll 1
  for (int kc = 0; kc < D_ / KC; ++kc) {
    __bf16* Xc = (kc & 1) ? Xs1 : Xs0;
    __bf16* Xn = (kc & 1) ? Xs0 : Xs1;
    if (kc < D_ / KC - 1) {                      // async-ish: issue next chunk's loads now
#pragma unroll
      for (int i = 0; i < 8; ++i) {
        const int f = tid + i * 256;
        xr[i] = *(const float4*)(x + (size_t)(tile + (f >> 5)) * D_ + (kc + 1) * KC + ((f & 31) << 2));
      }
    }
#pragma unroll
    for (int kk = 0; kk < KC / 32; ++kk) {
      const int kof = kk * 32;
      const int kg  = kc * KC + kof;
      const v16bf b0 = load_frag(w1e +  0 * D_ + kg, D_);
      const v16bf b1 = load_frag(w1e + 16 * D_ + kg, D_);
      const v16bf b2 = load_frag(w1e + 32 * D_ + kg, D_);
      const v16bf b3 = load_frag(w1e + 48 * D_ + kg, D_);
#pragma unroll
      for (int mt = 0; mt < 4; ++mt) {
        const v16bf a = load_frag(Xc + mt * 16 * XS_STRIDE + kof, XS_STRIDE);
        hacc[mt][0] = WMMA_BF16(a, b0, hacc[mt][0]);
        hacc[mt][1] = WMMA_BF16(a, b1, hacc[mt][1]);
        hacc[mt][2] = WMMA_BF16(a, b2, hacc[mt][2]);
        hacc[mt][3] = WMMA_BF16(a, b3, hacc[mt][3]);
      }
      if (wave < 4) {                            // wave-uniform; EXEC all-1 inside
        const v16bf bgf = load_frag(Wgbf + kg, D_);
        const v16bf ar  = load_frag(Xc + wave * 16 * XS_STRIDE + kof, XS_STRIDE);
        racc = WMMA_BF16(ar, bgf, racc);
      }
    }
    if (kc < D_ / KC - 1) {
#pragma unroll
      for (int i = 0; i < 8; ++i) {
        const int f = tid + i * 256;
        store4bf(Xn + (f >> 5) * XS_STRIDE + ((f & 31) << 2), xr[i]);
      }
    }
    __syncthreads();
  }

  // ---- router logits to LDS ----
  if (wave < 4 && (lane & 15) < 8) {
#pragma unroll
    for (int j = 0; j < 8; ++j) {
      const int row = wave * 16 + hi + j;
      Llog[row * E_ + (lane & 15)] = racc[j];
    }
  }
  __syncthreads();

  // ---- top-2 + softmax-over-selected, folded with SCALING=2.0 into dense gate table ----
  if (tid < MT) {
    float l[E_];
#pragma unroll
    for (int e = 0; e < E_; ++e) l[e] = Llog[tid * E_ + e] + bg[e];
    int i0 = 0; float m0 = l[0];
#pragma unroll
    for (int e = 1; e < E_; ++e) if (l[e] > m0) { m0 = l[e]; i0 = e; }
    int i1 = (i0 == 0) ? 1 : 0; float m1 = l[i1];
#pragma unroll
    for (int e = 0; e < E_; ++e) if (e != i0 && l[e] > m1) { m1 = l[e]; i1 = e; }
    const float p0 = 1.0f / (1.0f + __expf(m1 - m0));   // stable: m1 <= m0
#pragma unroll
    for (int e = 0; e < E_; ++e) Gt[tid * E_ + e] = 0.0f;
    Gt[tid * E_ + i0] = p0 * 2.0f;
    Gt[tid * E_ + i1] = (1.0f - p0) * 2.0f;
  }
  __syncthreads();

  // ---- gelu(h) * gate -> Hs[token][e*64+r] (bf16) ----
#pragma unroll
  for (int mt = 0; mt < 4; ++mt) {
#pragma unroll
    for (int nt = 0; nt < 4; ++nt) {
      const v8f v = hacc[mt][nt];
#pragma unroll
      for (int j = 0; j < 8; ++j) {
        const int row = mt * 16 + hi + j;
        const float g = Gt[row * E_ + wave];
        float h = v[j];
        h = 0.5f * h * (1.0f + erff(h * 0.70710678118654752f));  // exact GELU
        Hs[row * HS_STRIDE + wave * R_ + nt * 16 + (lane & 15)] = (__bf16)(h * g);
      }
    }
  }
  __syncthreads();

  // ---- phase 2: out = Hs @ W2r^T  (K = E*R = 512); wave owns a 128-col D strip ----
#pragma unroll 1
  for (int s = 0; s < 8; ++s) {
    const int d0 = wave * 128 + s * 16;
    v8f acc[4] = { vz, vz, vz, vz };
#pragma unroll
    for (int kk = 0; kk < 16; ++kk) {
      const v16bf b = load_frag(W2r + (size_t)d0 * HK + kk * 32, HK);
#pragma unroll
      for (int mt = 0; mt < 4; ++mt) {
        const v16bf a = load_frag(Hs + mt * 16 * HS_STRIDE + kk * 32, HS_STRIDE);
        acc[mt] = WMMA_BF16(a, b, acc[mt]);
      }
    }
#pragma unroll
    for (int mt = 0; mt < 4; ++mt)
#pragma unroll
      for (int j = 0; j < 8; ++j) {
        const int row = tile + mt * 16 + hi + j;
        out[(size_t)row * D_ + d0 + (lane & 15)] = acc[mt][j];
      }
  }
}

extern "C" void kernel_launch(void* const* d_in, const int* in_sizes, int n_in,
                              void* d_out, int out_size, void* d_ws, size_t ws_size,
                              hipStream_t stream) {
  const float* x  = (const float*)d_in[0];
  const float* Wg = (const float*)d_in[1];
  const float* bg = (const float*)d_in[2];
  const float* W1 = (const float*)d_in[3];
  const float* W2 = (const float*)d_in[4];
  float* out = (float*)d_out;

  // workspace layout (needs ~2.1 MB): W1bf | W2r | Wgbf
  __bf16* W1bf = (__bf16*)d_ws;
  __bf16* W2r  = W1bf + (size_t)E_ * R_ * D_;
  __bf16* Wgbf = W2r  + (size_t)D_ * HK;

  const int N = in_sizes[0] / D_;   // 16384 tokens

  moe_lora_prep<<<(E_ * R_ * D_ + 255) / 256, 256, 0, stream>>>(Wg, W1, W2, W1bf, W2r, Wgbf);

  const size_t shmem = (size_t)(2 * MT * XS_STRIDE + MT * HS_STRIDE) * sizeof(__bf16)
                     + (size_t)2 * MT * E_ * sizeof(float);   // ~103 KB (WGP has 320 KB LDS)
  (void)hipFuncSetAttribute((const void*)moe_lora_main,
                            hipFuncAttributeMaxDynamicSharedMemorySize, (int)shmem);
  moe_lora_main<<<N / MT, 256, shmem, stream>>>(x, bg, W1bf, W2r, Wgbf, out);
}
